// SlotEquivariantHead_76227079569885
// MI455X (gfx1250) — matrix operand order repaired and checked by reading, not compiled, not run on hardware
//
#include <hip/hip_runtime.h>
#include <hip/hip_bf16.h>

// ---------------------------------------------------------------------------
// SlotEquivariantHead: 16-step scan of a 3-layer MLP (162->256->256->96) over
// B=16384 rows, carry = last 32 outputs. Compute-bound (AI ~343 FLOP/B) ->
// matrix cores via v_wmma_f32_16x16x32_bf16 with f32 accumulation.
// One workgroup = 64 batch rows, 16 wave32 (4 waves/SIMD for latency hiding),
// loops p=0..15 with all weights resident in LDS (bf16, N-major so each
// B-fragment is two 16B ds loads per lane).
// ---------------------------------------------------------------------------

typedef __attribute__((ext_vector_type(16))) __bf16 v16bf;
typedef __attribute__((ext_vector_type(8)))  __bf16 v8bf;
typedef __attribute__((ext_vector_type(8)))  float  v8f;

#define Bsz   16384
#define Pn    16
#define Dn    64
#define Cn    32
#define Hn    256
#define BM    64          // batch rows per workgroup
#define NTHR  512         // 16 wave32

__device__ __forceinline__ v8f wmma_bf16(v16bf a, v16bf b, v8f c) {
  return __builtin_amdgcn_wmma_f32_16x16x32_bf16(
      /*neg_a=*/false, a, /*neg_b=*/false, b,
      /*c_mod=*/(short)0, c, /*reuse_a=*/false, /*reuse_b=*/false);
}

// A fragment (16x32 bf16) from a row-major LDS tile [rows][stride].
// Lane l (l<16): M=l, VGPR0-3 hold K=kc..kc+7, VGPR4-7 hold K=kc+16..kc+23.
// Lane l>=16 : same rows, K offset +8.  -> two contiguous 16B loads per lane.
__device__ __forceinline__ v16bf load_a_frag(const __bf16* tile, int stride,
                                             int mrow, int kc, int lm, int lh) {
  const __bf16* p = tile + (mrow + lm) * stride + kc + lh * 8;
  v8bf lo = *(const v8bf*)(p);
  v8bf hi = *(const v8bf*)(p + 16);
  union { v16bf v; v8bf h[2]; } u;
  u.h[0] = lo; u.h[1] = hi;
  return u.v;
}

// B fragment (32x16 bf16) from an N-major LDS weight [n][kstride].
// Lane l: N=l&15, K = kc+(l>>4)*8..+7 and kc+16+(l>>4)*8..+7 -> two 16B loads.
__device__ __forceinline__ v16bf load_b_frag(const __bf16* w, int kstride,
                                             int ncol, int kc, int lm, int lh) {
  const __bf16* p = w + (ncol + lm) * kstride + kc + lh * 8;
  union { v16bf v; v8bf h[2]; } u;
  u.h[0] = *(const v8bf*)(p); u.h[1] = *(const v8bf*)(p + 16);
  return u.v;
}

__device__ __forceinline__ void cvt_store8(__bf16* d, float4 a, float4 b) {
  d[0]=(__bf16)a.x; d[1]=(__bf16)a.y; d[2]=(__bf16)a.z; d[3]=(__bf16)a.w;
  d[4]=(__bf16)b.x; d[5]=(__bf16)b.y; d[6]=(__bf16)b.z; d[7]=(__bf16)b.w;
}

__global__ __launch_bounds__(NTHR, 1)
void slot_head_scan_kernel(const float* __restrict__ bias_a,
                           const float* __restrict__ bias_b,
                           const float* __restrict__ op,
                           const float* __restrict__ W1,
                           const float* __restrict__ b1,
                           const float* __restrict__ W2,
                           const float* __restrict__ b2,
                           const float* __restrict__ W3,
                           const float* __restrict__ b3,
                           float* __restrict__ out) {
  // ---- LDS: weights (bf16, N-major) + activation tiles -------------------
  __shared__ __bf16 sW1a[Hn * 64];    // W1 rows 0..63   (a part)   32 KB
  __shared__ __bf16 sW1b[Hn * 64];    // W1 rows 64..127 (b part)   32 KB
  __shared__ __bf16 sW1c[Hn * 32];    // W1 rows 130..161 (carry)   16 KB
  __shared__ __bf16 sW2 [Hn * Hn];    // W2                        128 KB
  __shared__ __bf16 sW3 [96 * Hn];    // W3                         48 KB
  __shared__ __bf16 sA  [BM * 64];    // a_p tile   8 KB
  __shared__ __bf16 sBt [BM * 64];    // b_p tile   8 KB
  __shared__ __bf16 sC  [BM * 32];    // carry      4 KB
  __shared__ __bf16 sH  [BM * Hn];    // h1/h2     32 KB
  __shared__ float  sB1[Hn], sB2[Hn], sB3[96];
  __shared__ float  sOp0w[Hn], sOp1w[Hn];   // W1 rows 128,129 (op part)
  __shared__ float  sOpv[BM * 2];           // per-row op values

  const int t    = threadIdx.x;
  const int lane = t & 31;
  const int lm   = lane & 15;
  const int lh   = lane >> 4;
  const int wid  = t >> 5;            // 0..15
  const int mrow = (wid >> 2) * 16;   // 4 M-tiles
  const int wcol = wid & 3;           // 4 N-columns of 64
  const int gm0  = blockIdx.x * BM;

  // ---- one-time weight convert f32 -> bf16, transposed to N-major --------
  for (int idx = t; idx < Hn * 64; idx += NTHR) {   // sW1a / sW1b
    int n = idx >> 6, k = idx & 63;
    sW1a[n * 64 + k] = (__bf16)W1[k * Hn + n];
    sW1b[n * 64 + k] = (__bf16)W1[(k + 64) * Hn + n];
  }
  for (int idx = t; idx < Hn * 32; idx += NTHR) {   // sW1c (rows 130..161)
    int n = idx >> 5, k = idx & 31;
    sW1c[n * 32 + k] = (__bf16)W1[(130 + k) * Hn + n];
  }
  for (int idx = t; idx < Hn * Hn; idx += NTHR) {   // sW2
    int n = idx >> 8, k = idx & 255;
    sW2[n * Hn + k] = (__bf16)W2[k * Hn + n];
  }
  for (int idx = t; idx < 96 * Hn; idx += NTHR) {   // sW3
    int n = idx >> 8, k = idx & 255;
    sW3[n * Hn + k] = (__bf16)W3[k * 96 + n];
  }
  if (t < Hn) {
    sB1[t]   = b1[t];
    sB2[t]   = b2[t];
    sOp0w[t] = W1[128 * Hn + t];
    sOp1w[t] = W1[129 * Hn + t];
  }
  if (t < 96) sB3[t] = b3[t];
  if (t < BM) {
    sOpv[t * 2]     = op[(gm0 + t) * 2];
    sOpv[t * 2 + 1] = op[(gm0 + t) * 2 + 1];
  }
  for (int idx = t; idx < BM * 32; idx += NTHR) sC[idx] = (__bf16)0.0f; // carry0
  __syncthreads();

  // ---- sequential scan over digit positions ------------------------------
  for (int p = 0; p < Pn; ++p) {
    // Phase A: stage a_p / b_p tiles (f32 load -> bf16 -> LDS)
    {
      const int row = t >> 3;           // 0..63
      const int ds  = (t & 7) * 8;      // 8 floats per thread per tensor
      const float* pa = bias_a + ((size_t)(gm0 + row) * Pn + p) * Dn + ds;
      const float* pb = bias_b + ((size_t)(gm0 + row) * Pn + p) * Dn + ds;
      const float4* pa4 = (const float4*)pa;
      const float4* pb4 = (const float4*)pb;
      float4 a0 = pa4[0], a1 = pa4[1];
      float4 c0 = pb4[0], c1 = pb4[1];
      cvt_store8(sA  + row * 64 + ds, a0, a1);
      cvt_store8(sBt + row * 64 + ds, c0, c1);
      if (p < Pn - 1) {   // prefetch next slot (global_prefetch_b8)
        __builtin_prefetch(pa + Dn, 0, 0);
        __builtin_prefetch(pb + Dn, 0, 0);
      }
    }
    __syncthreads();   // a/b staged; prev-step carry write + sH reads complete

    // GEMM1: h1 = relu(a@W1a + b@W1b + carry@W1c + op@W1op + b1)
    {
      v16bf fa0 = load_a_frag(sA , 64, mrow,  0, lm, lh);
      v16bf fa1 = load_a_frag(sA , 64, mrow, 32, lm, lh);
      v16bf fb0 = load_a_frag(sBt, 64, mrow,  0, lm, lh);
      v16bf fb1 = load_a_frag(sBt, 64, mrow, 32, lm, lh);
      v16bf fc0 = load_a_frag(sC , 32, mrow,  0, lm, lh);
      v8f acc[4];
#pragma unroll
      for (int j = 0; j < 4; ++j) {
        const int ncol = wcol * 64 + j * 16;
        const int n    = ncol + lm;
        const float w0 = sOp0w[n], w1 = sOp1w[n], bb = sB1[n];
        v8f a;
#pragma unroll
        for (int r = 0; r < 8; ++r) {
          const int m = mrow + lh * 8 + r;
          a[r] = bb + sOpv[m * 2] * w0 + sOpv[m * 2 + 1] * w1;
        }
        a = wmma_bf16(fa0, load_b_frag(sW1a, 64, ncol,  0, lm, lh), a);
        a = wmma_bf16(fa1, load_b_frag(sW1a, 64, ncol, 32, lm, lh), a);
        a = wmma_bf16(fb0, load_b_frag(sW1b, 64, ncol,  0, lm, lh), a);
        a = wmma_bf16(fb1, load_b_frag(sW1b, 64, ncol, 32, lm, lh), a);
        a = wmma_bf16(fc0, load_b_frag(sW1c, 32, ncol,  0, lm, lh), a);
        acc[j] = a;
      }
#pragma unroll
      for (int j = 0; j < 4; ++j) {
        const int n = wcol * 64 + j * 16 + lm;
#pragma unroll
        for (int r = 0; r < 8; ++r) {
          const int m = mrow + lh * 8 + r;
          sH[m * Hn + n] = (__bf16)fmaxf(acc[j][r], 0.0f);
        }
      }
    }
    __syncthreads();   // h1 visible

    // GEMM2: h2 = relu(h1@W2 + b2)   (read all of sH, then overwrite it)
    {
      v8f acc[4];
#pragma unroll
      for (int j = 0; j < 4; ++j) {
        const float bb = sB2[wcol * 64 + j * 16 + lm];
        v8f a;
#pragma unroll
        for (int r = 0; r < 8; ++r) a[r] = bb;
        acc[j] = a;
      }
#pragma unroll
      for (int kc = 0; kc < Hn; kc += 32) {
        v16bf ah = load_a_frag(sH, Hn, mrow, kc, lm, lh);
#pragma unroll
        for (int j = 0; j < 4; ++j)
          acc[j] = wmma_bf16(ah, load_b_frag(sW2, Hn, wcol * 64 + j * 16, kc, lm, lh), acc[j]);
      }
      __syncthreads();   // all h1 reads complete before overwriting sH
#pragma unroll
      for (int j = 0; j < 4; ++j) {
        const int n = wcol * 64 + j * 16 + lm;
#pragma unroll
        for (int r = 0; r < 8; ++r) {
          const int m = mrow + lh * 8 + r;
          sH[m * Hn + n] = (__bf16)fmaxf(acc[j][r], 0.0f);
        }
      }
    }
    __syncthreads();   // h2 visible

    // GEMM3: y = h2@W3 + b3 ; y[:, :64] -> out, y[:, 64:96] -> carry
    if (wcol < 2) {            // digit outputs (N = 0..63), 2 tiles per wave
      v8f acc[2];
#pragma unroll
      for (int j = 0; j < 2; ++j) {
        const float bb = sB3[wcol * 32 + j * 16 + lm];
        v8f a;
#pragma unroll
        for (int r = 0; r < 8; ++r) a[r] = bb;
        acc[j] = a;
      }
#pragma unroll
      for (int kc = 0; kc < Hn; kc += 32) {
        v16bf ah = load_a_frag(sH, Hn, mrow, kc, lm, lh);
#pragma unroll
        for (int j = 0; j < 2; ++j)
          acc[j] = wmma_bf16(ah, load_b_frag(sW3, Hn, wcol * 32 + j * 16, kc, lm, lh), acc[j]);
      }
#pragma unroll
      for (int j = 0; j < 2; ++j) {
        const int n = wcol * 32 + j * 16 + lm;
#pragma unroll
        for (int r = 0; r < 8; ++r) {
          const int m = mrow + lh * 8 + r;
          out[(size_t)(gm0 + m) * (Pn * Dn) + p * Dn + n] = acc[j][r];
        }
      }
    } else if (wcol == 2) {    // carry outputs (N = 64..95), 2 tiles
      v8f acc[2];
#pragma unroll
      for (int j = 0; j < 2; ++j) {
        const float bb = sB3[64 + j * 16 + lm];
        v8f a;
#pragma unroll
        for (int r = 0; r < 8; ++r) a[r] = bb;
        acc[j] = a;
      }
#pragma unroll
      for (int kc = 0; kc < Hn; kc += 32) {
        v16bf ah = load_a_frag(sH, Hn, mrow, kc, lm, lh);
#pragma unroll
        for (int j = 0; j < 2; ++j)
          acc[j] = wmma_bf16(ah, load_b_frag(sW3, Hn, 64 + j * 16, kc, lm, lh), acc[j]);
      }
#pragma unroll
      for (int j = 0; j < 2; ++j) {
        const int n = j * 16 + lm;
#pragma unroll
        for (int r = 0; r < 8; ++r) {
          const int m = mrow + lh * 8 + r;
          sC[m * 32 + n] = (__bf16)acc[j][r];
        }
      }
    }
    // wcol==3 idles through GEMM3; next-iteration barrier orders sC/sH hazards
  }
}

extern "C" void kernel_launch(void* const* d_in, const int* in_sizes, int n_in,
                              void* d_out, int out_size, void* d_ws, size_t ws_size,
                              hipStream_t stream) {
  (void)in_sizes; (void)n_in; (void)d_ws; (void)ws_size; (void)out_size;
  const float* bias_a = (const float*)d_in[0];
  const float* bias_b = (const float*)d_in[1];
  const float* op     = (const float*)d_in[2];
  const float* W1     = (const float*)d_in[3];
  const float* b1     = (const float*)d_in[4];
  const float* W2     = (const float*)d_in[5];
  const float* b2     = (const float*)d_in[6];
  const float* W3     = (const float*)d_in[7];
  const float* b3     = (const float*)d_in[8];
  float* out = (float*)d_out;

  dim3 grid(Bsz / BM);   // 256 workgroups, one 64-row batch tile each
  dim3 block(NTHR);      // 16 wave32 -> 4 waves per SIMD32
  slot_head_scan_kernel<<<grid, block, 0, stream>>>(
      bias_a, bias_b, op, W1, b1, W2, b2, W3, b3, out);
}
